// DenseGCN_73770358276184
// MI455X (gfx1250) — matrix operand order repaired and checked by reading, not compiled, not run on hardware
//
#include <hip/hip_runtime.h>
#include <hip/hip_bf16.h>
#include <math.h>

// ---------------------------------------------------------------------------
// DenseGCN forward for MI455X (gfx1250, wave32, WMMA).
//
// Shapes: B=4, C_IN=32, N=4096, DIM=64, k=20.
// out = concat(y[64], y1[192], y2[576], y3[1728]) = 2560 ch, fp32.
//
// Precision plan:
//  - KNN distances: fp32 VALU (discrete ranking output -> keep reference-exact path)
//  - GEMMs: f16 inputs, fp32 accumulate via v_wmma_f32_16x16x32_f16
//    (BN renormalizes right after each GEMM, so f16 operand rounding is benign)
//  - Conv bias is absorbed exactly by train-mode BN -> skipped.
// ---------------------------------------------------------------------------

#define BATCH 4
#define NPTS  4096
#define PTS   (BATCH * NPTS)     // 16384 points total
#define CIN   32
#define DIMC  64
#define KNN_K 20
#define OUTC  2560

typedef __attribute__((ext_vector_type(16))) _Float16 v16h;
typedef __attribute__((ext_vector_type(8)))  _Float16 v8h;
typedef __attribute__((ext_vector_type(8)))  float    v8f;

// -------------------------------------------------------------------- sqnorm
__global__ __launch_bounds__(256) void sqnorm_kernel(const float* __restrict__ x,
                                                     float* __restrict__ sq) {
  int p = blockIdx.x * 256 + threadIdx.x;          // p in [0, PTS)
  int b = p >> 12, n = p & (NPTS - 1);
  const float* xb = x + (size_t)b * CIN * NPTS + n;
  float s = 0.f;
#pragma unroll
  for (int c = 0; c < CIN; ++c) {
    float v = xb[c * NPTS];
    s = fmaf(v, v, s);
  }
  sq[p] = s;
}

// -------------------------------------------------------------------- KNN
// One thread per point-row. Candidate features tiled through LDS; all lanes
// walk the same candidate index -> LDS broadcast reads (conflict-free).
// Top-20 kept in registers via fully-unrolled insertion; strict '>' keeps the
// earliest index on ties, matching jax.lax.top_k ordering.
__global__ __launch_bounds__(256) void knn_kernel(const float* __restrict__ x,
                                                  const float* __restrict__ sq,
                                                  int* __restrict__ idx) {
  __shared__ float tile[CIN][257];   // padded: 32 x 256 candidate features
  __shared__ float tsq[256];

  int tid = threadIdx.x;
  int p = blockIdx.x * 256 + tid;
  int b = p >> 12, n = p & (NPTS - 1);
  const float* xb = x + (size_t)b * CIN * NPTS;

  float xn[CIN];
#pragma unroll
  for (int c = 0; c < CIN; ++c) xn[c] = xb[c * NPTS + n];
  float sqn = sq[p];

  float best[KNN_K];
  int bidx[KNN_K];
#pragma unroll
  for (int s = 0; s < KNN_K; ++s) { best[s] = -INFINITY; bidx[s] = 0; }

  for (int t = 0; t < NPTS / 256; ++t) {
    __syncthreads();
    int m0 = t * 256;
#pragma unroll
    for (int c = 0; c < CIN; ++c) tile[c][tid] = xb[c * NPTS + m0 + tid];
    tsq[tid] = sq[(b << 12) + m0 + tid];
    __syncthreads();

    for (int j = 0; j < 256; ++j) {
      float dot = 0.f;
#pragma unroll
      for (int c = 0; c < CIN; ++c) dot = fmaf(xn[c], tile[c][j], dot);
      float v = 2.f * dot - sqn - tsq[j];          // neg squared distance
      int m = m0 + j;
      if (v > best[KNN_K - 1]) {
#pragma unroll
        for (int s = KNN_K - 1; s >= 0; --s) {
          if (s == 0 || v <= best[s - 1]) { best[s] = v; bidx[s] = m; break; }
          best[s] = best[s - 1]; bidx[s] = bidx[s - 1];
        }
      }
    }
  }
#pragma unroll
  for (int s = 0; s < KNN_K; ++s) idx[(size_t)p * KNN_K + s] = bidx[s];
}

// -------------------------------------------------------- graph feature (y)
// y = [max_k(nbr - x) (32ch) ; x (32ch)].  Writes:
//  - fp32 out channels [0,64)
//  - f16 into actT1[p,0:64], actT2[p,0:64] (y1 head), actT3[p,384:448] (y in y2)
__global__ __launch_bounds__(256) void gfeat_kernel(const float* __restrict__ x,
                                                    const int* __restrict__ idx,
                                                    float* __restrict__ out,
                                                    _Float16* __restrict__ a1,
                                                    _Float16* __restrict__ a2,
                                                    _Float16* __restrict__ a3) {
  int p = blockIdx.x * 256 + threadIdx.x;
  int b = p >> 12, n = p & (NPTS - 1);
  const float* xb = x + (size_t)b * CIN * NPTS;

  float xo[CIN], mx[CIN];
#pragma unroll
  for (int c = 0; c < CIN; ++c) { xo[c] = xb[c * NPTS + n]; mx[c] = -INFINITY; }

  for (int s = 0; s < KNN_K; ++s) {
    int m = idx[(size_t)p * KNN_K + s];
#pragma unroll
    for (int c = 0; c < CIN; ++c)
      mx[c] = fmaxf(mx[c], xb[c * NPTS + m] - xo[c]);
  }

  float* ob = out + (size_t)b * OUTC * NPTS + n;
#pragma unroll
  for (int c = 0; c < DIMC; ++c) {
    float v = (c < CIN) ? mx[c] : xo[c - CIN];
    ob[(size_t)c * NPTS] = v;
    _Float16 h = (_Float16)v;
    a1[(size_t)p * 64 + c] = h;
    a2[(size_t)p * 192 + c] = h;
    a3[(size_t)p * 576 + 384 + c] = h;
  }
}

// ------------------------------------------------------------- f32 -> f16
__global__ __launch_bounds__(256) void cvt_f16_kernel(const float* __restrict__ src,
                                                      _Float16* __restrict__ dst,
                                                      int count) {
  int i = blockIdx.x * 256 + threadIdx.x;
  if (i < count) dst[i] = (_Float16)src[i];
}

// --------------------------------------------------------------- WMMA GEMM
// z[M x PTS] = W[M x K] * actT[PTS x K]^T, fp32 accumulation.
// grid = (PTS/128, M/64), 256 threads = 8 waves; wave -> 32x32 output tile
// (2x2 of 16x16 WMMA tiles, 4 v_wmma per 32-wide K step).
//
// Operand lane layouts (A/B: this kernel's contiguous k order, shared by both; C/D per ISA):
//  A (16x32 f16): lane L holds row M=L&15, K chunks [c0..c0+7] and
//                 [c0+8..c0+15] with c0 = 16*(L>>4)  -> two 16B loads.
//  B (32x16 f16): lane L holds col N=L&15, 16 contiguous K at base 16*(L>>4).
//  C/D (16x16 f32): VGPR r, lane L -> M = r + 8*(L>>4), N = L&15.
__global__ __launch_bounds__(256) void gemm_wmma_kernel(const _Float16* __restrict__ Wh,
                                                        const _Float16* __restrict__ act,
                                                        float* __restrict__ z,
                                                        int M, int K) {
  int lane = threadIdx.x & 31;
  int w    = threadIdx.x >> 5;
  int m0 = blockIdx.y * 64 + (w >> 2) * 32;
  int n0 = blockIdx.x * 128 + (w & 3) * 32;
  int lm = lane & 15;
  int hi = lane >> 4;

  const _Float16* A0p = Wh  + (size_t)(m0 + lm)      * K + hi * 16;
  const _Float16* A1p = Wh  + (size_t)(m0 + 16 + lm) * K + hi * 16;
  const _Float16* B0p = act + (size_t)(n0 + lm)      * K + hi * 16;
  const _Float16* B1p = act + (size_t)(n0 + 16 + lm) * K + hi * 16;

  v8f acc00 = {}, acc01 = {}, acc10 = {}, acc11 = {};

  for (int kk = 0; kk < K; kk += 32) {
    if (kk + 32 < K) __builtin_prefetch(B0p + kk + 32, 0, 1);

    v8h a0lo = *(const v8h*)(A0p + kk);
    v8h a0hi = *(const v8h*)(A0p + kk + 8);
    v8h a1lo = *(const v8h*)(A1p + kk);
    v8h a1hi = *(const v8h*)(A1p + kk + 8);
    v16h a0 = __builtin_shufflevector(a0lo, a0hi, 0,1,2,3,4,5,6,7,8,9,10,11,12,13,14,15);
    v16h a1 = __builtin_shufflevector(a1lo, a1hi, 0,1,2,3,4,5,6,7,8,9,10,11,12,13,14,15);
    v16h b0 = *(const v16h*)(B0p + kk);
    v16h b1 = *(const v16h*)(B1p + kk);

    acc00 = __builtin_amdgcn_wmma_f32_16x16x32_f16(false, a0, false, b0, (short)0, acc00, false, false);
    acc01 = __builtin_amdgcn_wmma_f32_16x16x32_f16(false, a0, false, b1, (short)0, acc01, false, false);
    acc10 = __builtin_amdgcn_wmma_f32_16x16x32_f16(false, a1, false, b0, (short)0, acc10, false, false);
    acc11 = __builtin_amdgcn_wmma_f32_16x16x32_f16(false, a1, false, b1, (short)0, acc11, false, false);
  }

  int rb = hi * 8;
#pragma unroll
  for (int r = 0; r < 8; ++r) {
    z[(size_t)(m0 + r + rb)      * PTS + n0 + lm]      = acc00[r];
    z[(size_t)(m0 + r + rb)      * PTS + n0 + 16 + lm] = acc01[r];
    z[(size_t)(m0 + 16 + r + rb) * PTS + n0 + lm]      = acc10[r];
    z[(size_t)(m0 + 16 + r + rb) * PTS + n0 + 16 + lm] = acc11[r];
  }
}

// ----------------------------------------------------------- BN statistics
__global__ __launch_bounds__(256) void bn_stats_kernel(const float* __restrict__ z,
                                                       float* __restrict__ mean,
                                                       float* __restrict__ rstd) {
  __shared__ float rs[256], rq[256];
  int o = blockIdx.x;
  const float* row = z + (size_t)o * PTS;
  float s = 0.f, q = 0.f;
  for (int i = threadIdx.x; i < PTS; i += 256) {
    float v = row[i];
    s += v;
    q = fmaf(v, v, q);
  }
  rs[threadIdx.x] = s; rq[threadIdx.x] = q;
  __syncthreads();
  for (int d = 128; d > 0; d >>= 1) {
    if (threadIdx.x < d) { rs[threadIdx.x] += rs[threadIdx.x + d]; rq[threadIdx.x] += rq[threadIdx.x + d]; }
    __syncthreads();
  }
  if (threadIdx.x == 0) {
    float mu  = rs[0] * (1.f / PTS);
    float var = rq[0] * (1.f / PTS) - mu * mu;
    mean[o] = mu;
    rstd[o] = rsqrtf(var + 1e-5f);
  }
}

// -------------------------------------------------- BN + exact GELU + store
// Writes fp32 into out at channel base; optional f16 copies into next-layer
// activation buffers (d1/d2 with leading dims ld1/ld2 and column offsets).
__global__ __launch_bounds__(256) void bn_gelu_kernel(const float* __restrict__ z,
                                                      const float* __restrict__ mean,
                                                      const float* __restrict__ rstd,
                                                      const float* __restrict__ g,
                                                      const float* __restrict__ be,
                                                      float* __restrict__ out, int outChBase,
                                                      _Float16* __restrict__ d1, int ld1, int off1,
                                                      _Float16* __restrict__ d2, int ld2, int off2) {
  size_t i = (size_t)blockIdx.x * 256 + threadIdx.x;   // i = o*PTS + p
  int o = (int)(i >> 14);                               // PTS = 2^14
  int p = (int)(i & (PTS - 1));
  float v  = z[i];
  float zn = (v - mean[o]) * rstd[o] * g[o] + be[o];
  float h  = 0.5f * zn * (1.f + erff(zn * 0.70710678118654752f));

  int b = p >> 12, n = p & (NPTS - 1);
  out[((size_t)b * OUTC + outChBase + o) * NPTS + n] = h;

  _Float16 hh = (_Float16)h;
  if (d1) d1[(size_t)p * ld1 + off1 + o] = hh;
  if (d2) d2[(size_t)p * ld2 + off2 + o] = hh;
}

// ------------------------------------------------- duplicated concat ranges
// out channel map (per batch):
//   [0,64)=y  [64,128)=y dup  [128,256)=h1  [256,640)=h2  [640,832)=y1 dup
//   [832,1984)=h3  [1984,2560)=y2 dup
// All duplicates resolve into canonical ranges [0,832).
__global__ __launch_bounds__(256) void replicate_kernel(float* __restrict__ out) {
  size_t i = (size_t)blockIdx.x * 256 + threadIdx.x;   // over BATCH*832*NPTS
  int n = (int)(i & (NPTS - 1));
  int r = (int)(i >> 12);
  int c = r % 832;
  int b = r / 832;
  int dst, src;
  if      (c <  64) { dst = 64   + c;         src = c;               }
  else if (c < 128) { dst = 640  + (c - 64);  src = c - 64;          }
  else if (c < 256) { dst = 704  + (c - 128); src = c;               } // h1
  else if (c < 640) { dst = 1984 + (c - 256); src = c;               } // h2
  else if (c < 704) { dst = 2368 + (c - 640); src = c - 640;         } // y
  else              { dst = 2432 + (c - 704); src = 128 + (c - 704); } // h1
  size_t base = (size_t)b * OUTC * NPTS + n;
  out[base + (size_t)dst * NPTS] = out[base + (size_t)src * NPTS];
}

// ---------------------------------------------------------------------------
extern "C" void kernel_launch(void* const* d_in, const int* in_sizes, int n_in,
                              void* d_out, int out_size, void* d_ws, size_t ws_size,
                              hipStream_t stream) {
  (void)in_sizes; (void)n_in; (void)out_size; (void)ws_size;

  const float* x  = (const float*)d_in[0];
  const float* W1 = (const float*)d_in[1];   // b1 = d_in[2] absorbed by BN
  const float* g1 = (const float*)d_in[3];
  const float* e1 = (const float*)d_in[4];
  const float* W2 = (const float*)d_in[5];   // b2 = d_in[6] absorbed by BN
  const float* g2 = (const float*)d_in[7];
  const float* e2 = (const float*)d_in[8];
  const float* W3 = (const float*)d_in[9];   // b3 = d_in[10] absorbed by BN
  const float* g3 = (const float*)d_in[11];
  const float* e3 = (const float*)d_in[12];
  float* out = (float*)d_out;

  // Workspace layout (all offsets 256B aligned), total ~106 MB.
  char* ws = (char*)d_ws;
  size_t off = 0;
  auto take = [&](size_t bytes) { char* p = ws + off; off = (off + bytes + 255) & ~(size_t)255; return p; };
  float*    sq   = (float*)   take((size_t)PTS * 4);
  int*      idx  = (int*)     take((size_t)PTS * KNN_K * 4);
  _Float16* a1   = (_Float16*)take((size_t)PTS * 64  * 2);
  _Float16* a2   = (_Float16*)take((size_t)PTS * 192 * 2);
  _Float16* a3   = (_Float16*)take((size_t)PTS * 576 * 2);
  _Float16* wh1  = (_Float16*)take((size_t)128  * 64  * 2);
  _Float16* wh2  = (_Float16*)take((size_t)384  * 192 * 2);
  _Float16* wh3  = (_Float16*)take((size_t)1152 * 576 * 2);
  float*    mean = (float*)   take((size_t)1152 * 4);
  float*    rstd = (float*)   take((size_t)1152 * 4);
  float*    zbuf = (float*)   take((size_t)1152 * PTS * 4);

  // 1) KNN + graph feature
  sqnorm_kernel<<<PTS / 256, 256, 0, stream>>>(x, sq);
  knn_kernel<<<PTS / 256, 256, 0, stream>>>(x, sq, idx);
  gfeat_kernel<<<PTS / 256, 256, 0, stream>>>(x, idx, out, a1, a2, a3);

  // 2) Weights -> f16
  cvt_f16_kernel<<<(128 * 64 + 255) / 256, 256, 0, stream>>>(W1, wh1, 128 * 64);
  cvt_f16_kernel<<<(384 * 192 + 255) / 256, 256, 0, stream>>>(W2, wh2, 384 * 192);
  cvt_f16_kernel<<<(1152 * 576 + 255) / 256, 256, 0, stream>>>(W3, wh3, 1152 * 576);

  dim3 blk(256);

  // 3) Block 1: h1 = GELU(BN(W1 @ y)), out ch [128,256); feeds a2[64:192], a3[448:576]
  gemm_wmma_kernel<<<dim3(PTS / 128, 128 / 64), blk, 0, stream>>>(wh1, a1, zbuf, 128, 64);
  bn_stats_kernel<<<128, blk, 0, stream>>>(zbuf, mean, rstd);
  bn_gelu_kernel<<<128 * (PTS / 256), blk, 0, stream>>>(
      zbuf, mean, rstd, g1, e1, out, 128, a2, 192, 64, a3, 576, 448);

  // 4) Block 2: h2 = GELU(BN(W2 @ y1)), out ch [256,640); feeds a3[0:384]
  gemm_wmma_kernel<<<dim3(PTS / 128, 384 / 64), blk, 0, stream>>>(wh2, a2, zbuf, 384, 192);
  bn_stats_kernel<<<384, blk, 0, stream>>>(zbuf, mean, rstd);
  bn_gelu_kernel<<<384 * (PTS / 256), blk, 0, stream>>>(
      zbuf, mean, rstd, g2, e2, out, 256, a3, 576, 0, (_Float16*)nullptr, 0, 0);

  // 5) Block 3: h3 = GELU(BN(W3 @ y2)), out ch [832,1984)
  gemm_wmma_kernel<<<dim3(PTS / 128, 1152 / 64), blk, 0, stream>>>(wh3, a3, zbuf, 1152, 576);
  bn_stats_kernel<<<1152, blk, 0, stream>>>(zbuf, mean, rstd);
  bn_gelu_kernel<<<1152 * (PTS / 256), blk, 0, stream>>>(
      zbuf, mean, rstd, g3, e3, out, 832, (_Float16*)nullptr, 0, 0, (_Float16*)nullptr, 0, 0);

  // 6) Fill duplicated concat channel ranges
  replicate_kernel<<<BATCH * 832 * (NPTS / 256), blk, 0, stream>>>(out);
}
